// MultiHeadGraphAttention_38010460570141
// MI455X (gfx1250) — compile-verified, matrix operand order checked by reading, not run on hardware
//
#include <hip/hip_runtime.h>

#define DMODEL 128
#define NHEADS 8
#define DHEAD  16

typedef __attribute__((ext_vector_type(16))) _Float16     v16h;
typedef __attribute__((ext_vector_type(8)))  float        v8f;
typedef __attribute__((ext_vector_type(4)))  unsigned int v4u;
typedef __attribute__((ext_vector_type(8)))  unsigned int v8u;

__device__ __forceinline__ v8f wmma_f16(v16h a, v16h b, v8f c) {
    // (neg_a, A, neg_b, B, c_mod, C, reuse_a, reuse_b)
    return __builtin_amdgcn_wmma_f32_16x16x32_f16(false, a, false, b, (short)0, c,
                                                  false, false);
}

// ---------------------------------------------------------------------------
// Tensor Data Mover: 2D f32 tile (rows x cols) from global -> LDS.
// D# per CDNA5 ISA ch.8: group0 = {count, lds_addr, global_addr, type=2},
// group1 = {data_size=4B, tensor_dim0/1, tile_dim0/1, dim0_stride}.
// VADDR2/VADDR3 omitted (2D tensor). Rows beyond rows_tensor read as zero.
// ---------------------------------------------------------------------------
__device__ __forceinline__ void tdm_load_2d_f32(unsigned lds_off,
                                                const float* gptr,
                                                unsigned rows_tensor,
                                                unsigned rows_tile,
                                                unsigned cols,
                                                unsigned stride_elems) {
    unsigned long long ga = (unsigned long long)(size_t)gptr;
    v4u g0;
    g0[0] = 1u;                                                   // count=1
    g0[1] = lds_off;                                              // lds_addr
    g0[2] = (unsigned)ga;                                         // global_addr lo
    g0[3] = ((unsigned)(ga >> 32) & 0x1FFFFFFu) | 0x80000000u;    // addr hi | type=2
    v8u g1;
    g1[0] = 0x00020000u;                                          // data_size = 4B
    g1[1] = (cols & 0xFFFFu) << 16;                               // tensor_dim0 lo16
    g1[2] = (cols >> 16) | ((rows_tensor & 0xFFFFu) << 16);       // dim0 hi | dim1 lo
    g1[3] = (rows_tensor >> 16) | ((cols & 0xFFFFu) << 16);       // dim1 hi | tile_dim0
    g1[4] = rows_tile & 0xFFFFu;                                  // tile_dim1 (tile_dim2=0)
    g1[5] = stride_elems;                                         // dim0_stride lo32
    g1[6] = 0u;
    g1[7] = 0u;
    asm volatile("tensor_load_to_lds %0, %1" :: "s"(g0), "s"(g1) : "memory");
}

__device__ __forceinline__ unsigned lds_byte_off(const void* p) {
    // generic pointer to LDS: low 32 bits are the in-LDS byte address
    return (unsigned)(size_t)p;
}

// ---------------------------------------------------------------------------
// Zero workspace accumulators (sums [N*128] and cnt [N]).
// ---------------------------------------------------------------------------
__global__ void zero_kernel(float* __restrict__ sums, float* __restrict__ cnt,
                            int n128, int n) {
    int i = blockIdx.x * blockDim.x + threadIdx.x;
    if (i < n128) sums[i] = 0.0f;
    if (i < n)    cnt[i]  = 0.0f;
}

// ---------------------------------------------------------------------------
// Fused Q/K/V projection: [N,128] x [128,128] (x3) via v_wmma_f32_16x16x32_f16.
// Block = 128 threads = 4 waves; each wave owns a 16-row stripe (block: 64 rows).
// x tile (64x128 f32) and each weight (128x128 f32) staged in LDS by the TDM.
// ---------------------------------------------------------------------------
__global__ __launch_bounds__(128)
void qkv_wmma_kernel(const float* __restrict__ x,
                     const float* __restrict__ wq,
                     const float* __restrict__ wk,
                     const float* __restrict__ wv,
                     _Float16* __restrict__ Qh,
                     _Float16* __restrict__ Kh,
                     _Float16* __restrict__ Vh,
                     int n) {
    __shared__ float lds[(64 + DMODEL) * DMODEL];   // 32KB x-tile + 64KB weight
    float* ldsX = lds;
    float* ldsW = lds + 64 * DMODEL;

    const int tid    = threadIdx.x;
    const int wave   = tid >> 5;
    const int lane   = tid & 31;
    const int laneLo = lane & 15;
    const int hi     = (lane >> 4) & 1;
    const int m0     = blockIdx.x * 64 + wave * 16;
    const int kbase  = hi * 8;

    const float* Ws[3] = {wq, wk, wv};
    _Float16*    Os[3] = {Qh, Kh, Vh};

    // Stage x tile + first weight via TDM (one wave issues; OOB rows -> 0).
    if (wave == 0) {
        int rem = n - blockIdx.x * 64;
        unsigned rows = (rem > 64) ? 64u : (unsigned)rem;
        tdm_load_2d_f32(lds_byte_off(ldsX),
                        x + (size_t)blockIdx.x * 64 * DMODEL,
                        rows, 64u, DMODEL, DMODEL);
        tdm_load_2d_f32(lds_byte_off(ldsW), wq, DMODEL, DMODEL, DMODEL, DMODEL);
        __builtin_amdgcn_s_wait_tensorcnt(0);
    }
    __syncthreads();

#pragma unroll
    for (int mat = 0; mat < 3; ++mat) {
        v8f acc[8] = {};
#pragma unroll
        for (int ks = 0; ks < 4; ++ks) {
            const int kk = ks * 32;
            // A fragment: row (wave*16 + laneLo), K = kk + kbase + {0..7,16..23}
            const float* ap = ldsX + (wave * 16 + laneLo) * DMODEL + kk + kbase;
            v16h a;
#pragma unroll
            for (int j = 0; j < 8; ++j) {
                a[j]     = (_Float16)ap[j];
                a[j + 8] = (_Float16)ap[16 + j];
            }
#pragma unroll
            for (int nt = 0; nt < 8; ++nt) {
                // B fragment: col (nt*16 + laneLo), K rows strided in LDS
                const float* bp = ldsW + (size_t)(kk + kbase) * DMODEL + nt * 16 + laneLo;
                v16h b;
#pragma unroll
                for (int j = 0; j < 8; ++j) {
                    b[j]     = (_Float16)bp[j * DMODEL];
                    b[j + 8] = (_Float16)bp[(16 + j) * DMODEL];
                }
                acc[nt] = wmma_f16(a, b, acc[nt]);
            }
        }
        // D layout: VGPR r -> M = m0 + r + hi*8 ; N = nt*16 + laneLo
        _Float16* o = Os[mat];
#pragma unroll
        for (int nt = 0; nt < 8; ++nt) {
#pragma unroll
            for (int r = 0; r < 8; ++r) {
                int row = m0 + r + hi * 8;
                if (row < n)
                    o[(size_t)row * DMODEL + nt * 16 + laneLo] = (_Float16)acc[nt][r];
            }
        }
        if (mat < 2) {
            __syncthreads();                   // all waves done reading ldsW
            if (wave == 0) {
                tdm_load_2d_f32(lds_byte_off(ldsW), Ws[mat + 1],
                                DMODEL, DMODEL, DMODEL, DMODEL);
                __builtin_amdgcn_s_wait_tensorcnt(0);
            }
            __syncthreads();
        }
    }
}

// ---------------------------------------------------------------------------
// Edge stage: one thread per (edge, head). Q/K/V f16 (38 MB) resident in the
// 192 MB L2; scatter via global_atomic_add_f32.
// ---------------------------------------------------------------------------
__global__ __launch_bounds__(256)
void edge_kernel(const long long* __restrict__ eidx,
                 const float* __restrict__ eattr,
                 const _Float16* __restrict__ Qh,
                 const _Float16* __restrict__ Kh,
                 const _Float16* __restrict__ Vh,
                 float* __restrict__ sums,
                 float* __restrict__ cnt,
                 long long E) {
    long long idx = (long long)blockIdx.x * blockDim.x + threadIdx.x;
    if (idx >= E * NHEADS) return;
    const long long e = idx >> 3;
    const int h = (int)(idx & 7);

    const int src = (int)eidx[e];
    const int dst = (int)eidx[E + e];

    const float4* ea = (const float4*)(eattr + e * 16);
    float4 e0 = ea[0], e1 = ea[1], e2 = ea[2], e3 = ea[3];
    float s = (e0.x + e0.y + e0.z + e0.w) + (e1.x + e1.y + e1.z + e1.w) +
              (e2.x + e2.y + e2.z + e2.w) + (e3.x + e3.y + e3.z + e3.w);
    float ew = 1.0f / (1.0f + __expf(-s));

    const _Float16* qp = Qh + (size_t)src * DMODEL + h * DHEAD;
    const _Float16* kp = Kh + (size_t)dst * DMODEL + h * DHEAD;
    const _Float16* vp = Vh + (size_t)dst * DMODEL + h * DHEAD;

    float sc = 0.0f;
#pragma unroll
    for (int j = 0; j < DHEAD; ++j) sc += (float)qp[j] * (float)kp[j];
    sc *= 0.25f * ew;   // SCALE = 1/sqrt(16)

    float* sp = sums + (size_t)dst * DMODEL + h * DHEAD;
#pragma unroll
    for (int j = 0; j < DHEAD; ++j) atomicAdd(sp + j, (float)vp[j] * sc);
    if (h == 0) atomicAdd(cnt + dst, 1.0f);
}

// ---------------------------------------------------------------------------
// Output projection: out = (sums / max(cnt,1)) @ w_o + b_o via WMMA f16;
// w_o staged by TDM.
// ---------------------------------------------------------------------------
__global__ __launch_bounds__(128)
void out_wmma_kernel(const float* __restrict__ sums,
                     const float* __restrict__ cnt,
                     const float* __restrict__ wo,
                     const float* __restrict__ bo,
                     float* __restrict__ out,
                     int n) {
    __shared__ float ldsW[DMODEL * DMODEL];   // 64 KB

    const int tid    = threadIdx.x;
    const int wave   = tid >> 5;
    const int lane   = tid & 31;
    const int laneLo = lane & 15;
    const int hi     = (lane >> 4) & 1;
    const int m0     = blockIdx.x * 64 + wave * 16;
    const int kbase  = hi * 8;

    if (wave == 0) {
        tdm_load_2d_f32(lds_byte_off(ldsW), wo, DMODEL, DMODEL, DMODEL, DMODEL);
        __builtin_amdgcn_s_wait_tensorcnt(0);
    }

    int arow = m0 + laneLo;
    if (arow >= n) arow = n - 1;
    const float rcp = 1.0f / fmaxf(cnt[arow], 1.0f);
    __syncthreads();

    v8f acc[8] = {};
#pragma unroll
    for (int ks = 0; ks < 4; ++ks) {
        const int kk = ks * 32;
        const float* ap = sums + (size_t)arow * DMODEL + kk + kbase;
        v16h a;
#pragma unroll
        for (int j = 0; j < 8; ++j) {
            a[j]     = (_Float16)(ap[j] * rcp);
            a[j + 8] = (_Float16)(ap[16 + j] * rcp);
        }
#pragma unroll
        for (int nt = 0; nt < 8; ++nt) {
            const float* bp = ldsW + (size_t)(kk + kbase) * DMODEL + nt * 16 + laneLo;
            v16h b;
#pragma unroll
            for (int j = 0; j < 8; ++j) {
                b[j]     = (_Float16)bp[j * DMODEL];
                b[j + 8] = (_Float16)bp[(16 + j) * DMODEL];
            }
            acc[nt] = wmma_f16(a, b, acc[nt]);
        }
    }
#pragma unroll
    for (int nt = 0; nt < 8; ++nt) {
        const int col = nt * 16 + laneLo;
        const float bias = bo[col];
#pragma unroll
        for (int r = 0; r < 8; ++r) {
            int row = m0 + r + hi * 8;
            if (row < n) out[(size_t)row * DMODEL + col] = acc[nt][r] + bias;
        }
    }
}

// ---------------------------------------------------------------------------
extern "C" void kernel_launch(void* const* d_in, const int* in_sizes, int n_in,
                              void* d_out, int out_size, void* d_ws, size_t ws_size,
                              hipStream_t stream) {
    const float*     x     = (const float*)d_in[0];
    const long long* eidx  = (const long long*)d_in[1];   // int64 edge_index [2,E]
    const float*     eattr = (const float*)d_in[2];
    const float*     wq    = (const float*)d_in[3];
    const float*     wk    = (const float*)d_in[4];
    const float*     wv    = (const float*)d_in[5];
    const float*     wo    = (const float*)d_in[6];
    const float*     bo    = (const float*)d_in[7];
    float*           out   = (float*)d_out;

    const int       N = in_sizes[0] / DMODEL;
    const long long E = in_sizes[1] / 2;

    // Workspace carving: Qh,Kh,Vh (f16), sums (f32), cnt (f32)
    char* ws = (char*)d_ws;
    _Float16* Qh   = (_Float16*)ws;
    _Float16* Kh   = Qh + (size_t)N * DMODEL;
    _Float16* Vh   = Kh + (size_t)N * DMODEL;
    float*    sums = (float*)(Vh + (size_t)N * DMODEL);
    float*    cnt  = sums + (size_t)N * DMODEL;

    const int n128 = N * DMODEL;

    zero_kernel<<<(n128 + 255) / 256, 256, 0, stream>>>(sums, cnt, n128, N);

    qkv_wmma_kernel<<<(N + 63) / 64, 128, 0, stream>>>(x, wq, wk, wv, Qh, Kh, Vh, N);

    const long long total = E * NHEADS;
    edge_kernel<<<(int)((total + 255) / 256), 256, 0, stream>>>(
        eidx, eattr, Qh, Kh, Vh, sums, cnt, E);

    out_wmma_kernel<<<(N + 63) / 64, 128, 0, stream>>>(sums, cnt, wo, bo, out, N);
}
